// FusedGAT_30846455120744
// MI455X (gfx1250) — matrix operand order repaired and checked by reading, not compiled, not run on hardware
//
#include <hip/hip_runtime.h>
#include <hip/hip_bf16.h>

#define GN 50000
#define GE 1600000

typedef __attribute__((ext_vector_type(16))) _Float16 v16h;
typedef __attribute__((ext_vector_type(8)))  float    v8f;

union ABPack { v16h v; unsigned u[8]; };

// ---------------------------------------------------------------------------
// Elementwise helpers
// ---------------------------------------------------------------------------
__global__ void fill_u32_kernel(unsigned* __restrict__ p, unsigned val, long long n) {
    long long i = (long long)blockIdx.x * blockDim.x + threadIdx.x;
    long long stride = (long long)gridDim.x * blockDim.x;
    for (; i < n; i += stride) p[i] = val;
}

__global__ void cvt_f32_f16_kernel(const float* __restrict__ x, _Float16* __restrict__ xh, long long n) {
    long long i = (long long)blockIdx.x * blockDim.x + threadIdx.x;
    long long stride = (long long)gridDim.x * blockDim.x;
    for (; i < n; i += stride) xh[i] = (_Float16)x[i];
}

// Wt[n][k] = (f16) W[k][n]   (128x128)
__global__ void cvt_w_transpose_kernel(const float* __restrict__ W, _Float16* __restrict__ Wt) {
    int i = blockIdx.x * blockDim.x + threadIdx.x;
    if (i < 128 * 128) {
        int k = i >> 7, n = i & 127;
        Wt[n * 128 + k] = (_Float16)W[i];
    }
}

// xh[i] = (f16) relu(agg[i] + b[i%128])
__global__ void bias_relu_cvt_kernel(const float* __restrict__ agg, const float* __restrict__ b,
                                     _Float16* __restrict__ xh, long long n) {
    long long i = (long long)blockIdx.x * blockDim.x + threadIdx.x;
    long long stride = (long long)gridDim.x * blockDim.x;
    for (; i < n; i += stride) {
        float v = agg[i] + b[(int)(i & 127)];
        xh[i] = (_Float16)fmaxf(v, 0.0f);
    }
}

// ---------------------------------------------------------------------------
// WMMA GEMM: C[M x 128] = A[M x 128](f16 row-major) * W[128 x 128]
// (B given pre-transposed: Bt[n][k], f16). One 16x16 tile per wave, 8 waves
// per block cover all 128 output columns for a 16-row stripe.
// ---------------------------------------------------------------------------
__global__ void gemm_wmma_f16_kernel(const _Float16* __restrict__ A,
                                     const _Float16* __restrict__ Bt,
                                     float* __restrict__ C, int M) {
    const int lane = threadIdx.x & 31;
    const int wave = threadIdx.x >> 5;      // 0..7 -> column tile
    const int m0   = blockIdx.x * 16;
    const int n0   = wave * 16;
    const int half = lane >> 4;             // lane group 0/1
    const int r    = lane & 15;

    const unsigned* __restrict__ arow = (const unsigned*)(A  + (size_t)(m0 + r) * 128);
    const unsigned* __restrict__ brow = (const unsigned*)(Bt + (size_t)(n0 + r) * 128);

    v8f c = {};
#pragma unroll
    for (int kk = 0; kk < 128; kk += 32) {
        const int kw = kk >> 1;  // dword base index
        ABPack a, b;
#pragma unroll
        for (int j = 0; j < 4; ++j) {
            // A 16x32 f16 layout: VGPR j   : K = half*8 + 2j .. +1
            //                     VGPR j+4 : K = 16 + half*8 + 2j .. +1
            a.u[j]     = arow[kw + half * 4 + j];
            a.u[j + 4] = arow[kw + 8 + half * 4 + j];
            // B 32x16 f16 layout: same K mapping as A (VGPR j: half*8+2j, VGPR j+4: 16+half*8+2j)
            b.u[j]     = brow[kw + half * 4 + j];
            b.u[j + 4] = brow[kw + 8 + half * 4 + j];
        }
        c = __builtin_amdgcn_wmma_f32_16x16x32_f16(
                /*neg_a=*/false, a.v, /*neg_b=*/false, b.v,
                /*c_mod=*/(short)0, c, /*reuse_a=*/false, /*reuse_b=*/false);
    }

    // D layout: lanes 0-15: N=lane, M=r_vgpr; lanes 16-31: N=lane-16, M=8+r_vgpr
    float* __restrict__ crow = C + (size_t)(m0 + half * 8) * 128 + n0 + r;
#pragma unroll
    for (int rr = 0; rr < 8; ++rr) crow[(size_t)rr * 128] = c[rr];
}

// ---------------------------------------------------------------------------
// Per-row attention dot products: als[n,h] = <h[n,h,:], a_src[h,:]>, same for
// a_dst. Wave per row, G = 32/H lanes per head, shuffle reduction.
// ---------------------------------------------------------------------------
__global__ void alpha_dots_kernel(const float* __restrict__ h,
                                  const float* __restrict__ asrc,
                                  const float* __restrict__ adst,
                                  float* __restrict__ als, float* __restrict__ ald,
                                  int Nn, int H, int G) {
    int gwave  = (blockIdx.x * blockDim.x + threadIdx.x) >> 5;
    int lane   = threadIdx.x & 31;
    int nwaves = (gridDim.x * blockDim.x) >> 5;
    for (int n = gwave; n < Nn; n += nwaves) {
        int c4 = lane * 4;
        const float4 hv = *(const float4*)(h + (size_t)n * 128 + c4);
        const float4 as = *(const float4*)(asrc + c4);
        const float4 ad = *(const float4*)(adst + c4);
        float ps = hv.x * as.x + hv.y * as.y + hv.z * as.z + hv.w * as.w;
        float pd = hv.x * ad.x + hv.y * ad.y + hv.z * ad.z + hv.w * ad.w;
        for (int m = 1; m < G; m <<= 1) {
            ps += __shfl_xor(ps, m, 32);
            pd += __shfl_xor(pd, m, 32);
        }
        if ((lane & (G - 1)) == 0) {
            int hd = lane / G;
            als[(size_t)n * H + hd] = ps;
            ald[(size_t)n * H + hd] = pd;
        }
    }
}

// ---------------------------------------------------------------------------
// Edge pass 1: segment-max of leaky_relu(als[src]+ald[dst]) into order-
// preserving int encoding (L2-resident, atomicMax).
// ---------------------------------------------------------------------------
__device__ __forceinline__ float leaky02(float v) { return v > 0.0f ? v : 0.2f * v; }
__device__ __forceinline__ int enc_f32(float v) {
    int b = __float_as_int(v);
    return b >= 0 ? b : (b ^ 0x7fffffff);
}

__global__ void edge_max_kernel(const int* __restrict__ src, const int* __restrict__ dst,
                                const float* __restrict__ als, const float* __restrict__ ald,
                                int* __restrict__ menc, int E, int hshift) {
    const int H = 1 << hshift;
    long long total  = (long long)E << hshift;
    long long i      = (long long)blockIdx.x * blockDim.x + threadIdx.x;
    long long stride = (long long)gridDim.x * blockDim.x;
    for (; i < total; i += stride) {
        int e  = (int)(i >> hshift);
        int hd = (int)(i & (H - 1));
        int s = src[e], d = dst[e];
        float v = leaky02(als[(size_t)s * H + hd] + ald[(size_t)d * H + hd]);
        atomicMax(menc + (size_t)d * H + hd, enc_f32(v));
    }
}

__global__ void decode_max_kernel(const int* __restrict__ menc, float* __restrict__ mfl, long long n) {
    long long i = (long long)blockIdx.x * blockDim.x + threadIdx.x;
    long long stride = (long long)gridDim.x * blockDim.x;
    for (; i < n; i += stride) {
        int e = menc[i];
        float v = 0.0f;
        if (e != (int)0x80000000) {
            int b = e >= 0 ? e : (e ^ 0x7fffffff);
            v = __int_as_float(b);
        }
        mfl[i] = v;
    }
}

// Edge pass 2: denom[dst,h] += exp(e - m[dst,h])
__global__ void edge_sum_kernel(const int* __restrict__ src, const int* __restrict__ dst,
                                const float* __restrict__ als, const float* __restrict__ ald,
                                const float* __restrict__ mfl, float* __restrict__ den,
                                int E, int hshift) {
    const int H = 1 << hshift;
    long long total  = (long long)E << hshift;
    long long i      = (long long)blockIdx.x * blockDim.x + threadIdx.x;
    long long stride = (long long)gridDim.x * blockDim.x;
    for (; i < total; i += stride) {
        int e  = (int)(i >> hshift);
        int hd = (int)(i & (H - 1));
        int s = src[e], d = dst[e];
        float v = leaky02(als[(size_t)s * H + hd] + ald[(size_t)d * H + hd]);
        float p = __expf(v - mfl[(size_t)d * H + hd]);
        atomicAdd(den + (size_t)d * H + hd, p);
    }
}

// Edge pass 3 (bandwidth-dominant): out[dst] += h[src] * w. Wave per edge:
// lane handles 4 consecutive features (float4 gather + 4 L2 atomics).
__global__ void edge_agg_kernel(const int* __restrict__ src, const int* __restrict__ dst,
                                const float* __restrict__ als, const float* __restrict__ ald,
                                const float* __restrict__ mfl, const float* __restrict__ den,
                                const float* __restrict__ hfeat, float* __restrict__ out,
                                int E, int H, int dshift /*log2 of head dim*/) {
    int wave   = (blockIdx.x * blockDim.x + threadIdx.x) >> 5;
    int lane   = threadIdx.x & 31;
    int nwaves = (gridDim.x * blockDim.x) >> 5;
    int c4 = lane * 4;
    int hd = c4 >> dshift;
    for (int e = wave; e < E; e += nwaves) {
        int s = src[e], d = dst[e];
        float a = leaky02(als[(size_t)s * H + hd] + ald[(size_t)d * H + hd]);
        float w = __expf(a - mfl[(size_t)d * H + hd]) / (den[(size_t)d * H + hd] + 1e-16f);
        const float4 hv = *(const float4*)(hfeat + (size_t)s * 128 + c4);
        float* op = out + (size_t)d * 128 + c4;
        atomicAdd(op + 0, hv.x * w);
        atomicAdd(op + 1, hv.y * w);
        atomicAdd(op + 2, hv.z * w);
        atomicAdd(op + 3, hv.w * w);
    }
}

// ---------------------------------------------------------------------------
// log_softmax over 128 columns (adds b2 first); wave per row, in place.
// ---------------------------------------------------------------------------
__global__ void logsoftmax_kernel(float* __restrict__ out, const float* __restrict__ b2, int Nn) {
    int gwave  = (blockIdx.x * blockDim.x + threadIdx.x) >> 5;
    int lane   = threadIdx.x & 31;
    int nwaves = (gridDim.x * blockDim.x) >> 5;
    for (int n = gwave; n < Nn; n += nwaves) {
        int c4 = lane * 4;
        float4 v = *(const float4*)(out + (size_t)n * 128 + c4);
        const float4 b = *(const float4*)(b2 + c4);
        v.x += b.x; v.y += b.y; v.z += b.z; v.w += b.w;
        float mx = fmaxf(fmaxf(v.x, v.y), fmaxf(v.z, v.w));
        for (int m = 1; m < 32; m <<= 1) mx = fmaxf(mx, __shfl_xor(mx, m, 32));
        float sm = __expf(v.x - mx) + __expf(v.y - mx) + __expf(v.z - mx) + __expf(v.w - mx);
        for (int m = 1; m < 32; m <<= 1) sm += __shfl_xor(sm, m, 32);
        float lse = mx + logf(sm);
        float4 r = { v.x - lse, v.y - lse, v.z - lse, v.w - lse };
        *(float4*)(out + (size_t)n * 128 + c4) = r;
    }
}

// ---------------------------------------------------------------------------
// Launch
// ---------------------------------------------------------------------------
static inline int cdiv(long long a, int b) { return (int)((a + b - 1) / b); }

extern "C" void kernel_launch(void* const* d_in, const int* in_sizes, int n_in,
                              void* d_out, int out_size, void* d_ws, size_t ws_size,
                              hipStream_t stream) {
    const float* x     = (const float*)d_in[0];
    const int*   ei    = (const int*)d_in[1];
    const float* W1    = (const float*)d_in[2];
    const float* asrc1 = (const float*)d_in[3];
    const float* adst1 = (const float*)d_in[4];
    const float* b1    = (const float*)d_in[5];
    const float* W2    = (const float*)d_in[6];
    const float* asrc2 = (const float*)d_in[7];
    const float* adst2 = (const float*)d_in[8];
    const float* b2    = (const float*)d_in[9];
    float* out = (float*)d_out;

    const int*   srcv = ei;
    const int*   dstv = ei + GE;

    // workspace carve-up
    char* w = (char*)d_ws;
    _Float16* xh   = (_Float16*)w; w += (size_t)GN * 128 * sizeof(_Float16); // f16 features
    _Float16* Wt   = (_Float16*)w; w += (size_t)128 * 128 * sizeof(_Float16); // f16 W^T
    float*    hbuf = (float*)w;    w += (size_t)GN * 128 * sizeof(float);     // h = x@W
    float*    agg1 = (float*)w;    w += (size_t)GN * 128 * sizeof(float);     // layer1 aggregate
    float*    als  = (float*)w;    w += (size_t)GN * 4 * sizeof(float);
    float*    ald  = (float*)w;    w += (size_t)GN * 4 * sizeof(float);
    int*      menc = (int*)w;      w += (size_t)GN * 4 * sizeof(int);
    float*    mfl  = (float*)w;    w += (size_t)GN * 4 * sizeof(float);
    float*    den  = (float*)w;    w += (size_t)GN * 4 * sizeof(float);

    const int B = 256;
    const long long NF = (long long)GN * 128;

    // ---------------- Layer 1 ----------------
    cvt_f32_f16_kernel<<<cdiv(NF, B), B, 0, stream>>>(x, xh, NF);
    cvt_w_transpose_kernel<<<cdiv(128 * 128, B), B, 0, stream>>>(W1, Wt);
    gemm_wmma_f16_kernel<<<GN / 16, B, 0, stream>>>(xh, Wt, hbuf, GN);
    alpha_dots_kernel<<<cdiv(GN, 8), B, 0, stream>>>(hbuf, asrc1, adst1, als, ald, GN, 4, 8);

    fill_u32_kernel<<<cdiv((long long)GN * 4, B), B, 0, stream>>>((unsigned*)menc, 0x80000000u, (long long)GN * 4);
    fill_u32_kernel<<<cdiv((long long)GN * 4, B), B, 0, stream>>>((unsigned*)den, 0u, (long long)GN * 4);
    fill_u32_kernel<<<cdiv(NF, B), B, 0, stream>>>((unsigned*)agg1, 0u, NF);

    edge_max_kernel<<<cdiv((long long)GE * 4, B), B, 0, stream>>>(srcv, dstv, als, ald, menc, GE, 2);
    decode_max_kernel<<<cdiv((long long)GN * 4, B), B, 0, stream>>>(menc, mfl, (long long)GN * 4);
    edge_sum_kernel<<<cdiv((long long)GE * 4, B), B, 0, stream>>>(srcv, dstv, als, ald, mfl, den, GE, 2);
    edge_agg_kernel<<<16384, B, 0, stream>>>(srcv, dstv, als, ald, mfl, den, hbuf, agg1, GE, 4, 5);

    // relu(agg + b1) -> f16 features for layer 2
    bias_relu_cvt_kernel<<<cdiv(NF, B), B, 0, stream>>>(agg1, b1, xh, NF);

    // ---------------- Layer 2 ----------------
    cvt_w_transpose_kernel<<<cdiv(128 * 128, B), B, 0, stream>>>(W2, Wt);
    gemm_wmma_f16_kernel<<<GN / 16, B, 0, stream>>>(xh, Wt, hbuf, GN);
    alpha_dots_kernel<<<cdiv(GN, 8), B, 0, stream>>>(hbuf, asrc2, adst2, als, ald, GN, 1, 32);

    fill_u32_kernel<<<cdiv((long long)GN, B), B, 0, stream>>>((unsigned*)menc, 0x80000000u, (long long)GN);
    fill_u32_kernel<<<cdiv((long long)GN, B), B, 0, stream>>>((unsigned*)den, 0u, (long long)GN);
    fill_u32_kernel<<<cdiv(NF, B), B, 0, stream>>>((unsigned*)out, 0u, NF);

    edge_max_kernel<<<cdiv((long long)GE, B), B, 0, stream>>>(srcv, dstv, als, ald, menc, GE, 0);
    decode_max_kernel<<<cdiv((long long)GN, B), B, 0, stream>>>(menc, mfl, (long long)GN);
    edge_sum_kernel<<<cdiv((long long)GE, B), B, 0, stream>>>(srcv, dstv, als, ald, mfl, den, GE, 0);
    edge_agg_kernel<<<16384, B, 0, stream>>>(srcv, dstv, als, ald, mfl, den, hbuf, out, GE, 1, 7);

    logsoftmax_kernel<<<cdiv(GN, 8), B, 0, stream>>>(out, b2, GN);
}
